// HMLC_71734543778538
// MI455X (gfx1250) — compile-verified, matrix operand order checked by reading, not run on hardware
//
#include <hip/hip_runtime.h>

#define B_ANCH 1024
#define Q_SZ   32768
#define D_DIM  1024
#define L_LAB  4
#define INV_TEMP (1.0f / 0.07f)

typedef __attribute__((ext_vector_type(16))) __bf16 v16bf;
typedef __attribute__((ext_vector_type(8)))  float  v8f;
typedef __attribute__((ext_vector_type(4)))  int    v4i;
typedef unsigned short u16;

// ---- CDNA5 async-to-LDS availability probes ----
#if defined(__has_builtin)
# if __has_builtin(__builtin_amdgcn_global_load_async_to_lds_b128)
#  define HAVE_ASYNC_LDS 1
# endif
# if __has_builtin(__builtin_amdgcn_s_wait_asynccnt)
#  define HAVE_WAIT_ASYNC 1
# endif
#endif
#ifndef HAVE_ASYNC_LDS
# define HAVE_ASYNC_LDS 0
#endif
#ifndef HAVE_WAIT_ASYNC
# define HAVE_WAIT_ASYNC 0
#endif

// builtin signature (from hipcc diagnostics): (v4i addrspace(1)*, v4i addrspace(3)*, imm, imm)
typedef __attribute__((address_space(1))) v4i gv4i;
typedef __attribute__((address_space(3))) v4i lv4i;

// LDS staging geometry: per k-step slab = 512 rows x 32 bf16 (64 B/row),
// padded to 80 B row stride so ds_load_b128 fragment reads are conflict-free.
#define LDS_ROW_B 80
#define SLAB_B    (512 * LDS_ROW_B)   // 40960 B, double buffered -> 81920 B

// ---------- helpers ----------

__device__ __forceinline__ unsigned int pack2_bf16(float a, float b) {
  unsigned int ua = __float_as_uint(a);
  unsigned int ub = __float_as_uint(b);
  ua = ua + 0x7FFFu + ((ua >> 16) & 1u);   // round-to-nearest-even
  ub = ub + 0x7FFFu + ((ub >> 16) & 1u);
  return (ua >> 16) | (ub & 0xFFFF0000u);
}

// ---------- kernel 0: bulk f32 -> bf16 conversion (once, b128 in/out) ----------

__global__ __launch_bounds__(256) void hmlc_cvt(const float* __restrict__ src,
                                                u16* __restrict__ dst, size_t n) {
  size_t i = ((size_t)blockIdx.x * blockDim.x + threadIdx.x) * 8;
  if (i >= n) return;
  const float4* s = (const float4*)(src + i);
  float4 a = s[0], b = s[1];
  uint4 o;
  o.x = pack2_bf16(a.x, a.y);
  o.y = pack2_bf16(a.z, a.w);
  o.z = pack2_bf16(b.x, b.y);
  o.w = pack2_bf16(b.z, b.w);
  *(uint4*)(dst + i) = o;
}

// ---------- bf16 fragment loads ----------
// 16-bit A/B layout (wave32): lanes 0-15 = rows, K {0..7,16..23};
// lanes 16-31 = same rows, K {8..15,24..31}.

__device__ __forceinline__ v16bf load_frag_g(const u16* __restrict__ base, int k0, int lane) {
  const int r  = lane & 15;
  const int kh = (lane >> 4) & 1;
  const u16* p = base + (size_t)r * D_DIM + k0 + kh * 8;
  union { v16bf v; uint4 q[2]; } f;
  f.q[0] = *(const uint4*)p;
  f.q[1] = *(const uint4*)(p + 16);
  return f.v;
}

__device__ __forceinline__ v16bf load_frag_lds(const char* __restrict__ buf, int t, int lane) {
  const int r  = t * 16 + (lane & 15);
  const int kh = (lane >> 4) & 1;
  const char* p = buf + r * LDS_ROW_B + kh * 16;
  union { v16bf v; uint4 q[2]; } f;
  f.q[0] = *(const uint4*)p;
  f.q[1] = *(const uint4*)(p + 32);
  return f.v;
}

// Stage a 512x32 bf16 B-slab (rows n0..n0+511, K k0..k0+31) into LDS.
// 2048 x 16B chunks; 256 threads x 8 chunks.
__device__ __forceinline__ void stage_b_slab(const u16* __restrict__ gbase /* row0 @ k0 */,
                                             char* __restrict__ lbuf, int tid) {
#pragma unroll
  for (int j = 0; j < 8; ++j) {
    const int c    = tid + j * 256;     // 0..2047
    const int r    = c >> 2;
    const int part = c & 3;
    const u16* g = gbase + (size_t)r * D_DIM + part * 8;
    char*      l = lbuf + r * LDS_ROW_B + part * 16;
#if HAVE_ASYNC_LDS
    __builtin_amdgcn_global_load_async_to_lds_b128((gv4i*)g, (lv4i*)l, 0, 0);
#else
    *(uint4*)l = *(const uint4*)g;
#endif
  }
}

// ---------- kernel 1: init masks/scalars ----------

__global__ void hmlc_init(int* qa, int* aa, int* order, int* base_loc, float* scal) {
  int idx = blockIdx.x * blockDim.x + threadIdx.x;
  if (idx < Q_SZ) qa[idx] = 1;
  if (idx < B_ANCH) { aa[idx] = 1; order[idx] = idx; }
  if (idx == 0) { *base_loc = 0; scal[0] = 0.0f; scal[1] = -INFINITY; }
}

// ---------- kernel 2: base = max(label) ----------

__global__ void hmlc_base(const int* __restrict__ labels,
                          const int* __restrict__ labels_q, int* base_loc) {
  int idx = blockIdx.x * blockDim.x + threadIdx.x;
  const int nb = B_ANCH * L_LAB;
  const int tot = nb + Q_SZ * L_LAB;
  if (idx >= tot) return;
  int v = (idx < nb) ? labels[idx] : labels_q[idx - nb];
  atomicMax(base_loc, v);
}

// ---------- kernel 3: sim = (A @ Bq^T) / TEMP via WMMA bf16 ----------
// Block: 8 waves, tile 32(M) x 512(N). B slab double-buffered in LDS via
// async copies; compute on slab k overlaps DMA of slab k+1.

__global__ __launch_bounds__(256) void hmlc_gemm(const u16* __restrict__ Abf,
                                                 const u16* __restrict__ Bbf,
                                                 float* __restrict__ sim) {
  extern __shared__ char lds[];             // 2 * SLAB_B
  const int tid  = threadIdx.x;
  const int lane = tid & 31;
  const int wave = tid >> 5;                // 0..7
  const int wm = wave >> 2;                 // 0..1
  const int wn = wave & 3;                  // 0..3
  const int m0 = blockIdx.y * 32 + wm * 16;
  const int n0 = blockIdx.x * 512;          // block's B row base
  const int tb = wn * 8;                    // this wave's first of 8 B tiles in the slab

  const u16* At = Abf + (size_t)m0 * D_DIM;
  const u16* Bt = Bbf + (size_t)n0 * D_DIM;

  v8f acc[8] = {};
  const int NK = D_DIM / 32;

  stage_b_slab(Bt, lds, tid);               // prologue: slab 0 -> buf 0

  for (int it = 0; it < NK; ++it) {
    const int k0 = it * 32;
    char* cur = lds + (it & 1) * SLAB_B;
    char* nxt = lds + ((it + 1) & 1) * SLAB_B;
    if (it + 1 < NK) {
      stage_b_slab(Bt + (k0 + 32), nxt, tid);
#if HAVE_ASYNC_LDS
# if HAVE_WAIT_ASYNC
      __builtin_amdgcn_s_wait_asynccnt(8);  // my 'cur' copies done; 'nxt' in flight
# else
      asm volatile("s_wait_asynccnt 8" ::: "memory");
# endif
#endif
    } else {
#if HAVE_ASYNC_LDS
# if HAVE_WAIT_ASYNC
      __builtin_amdgcn_s_wait_asynccnt(0);  // drain
# else
      asm volatile("s_wait_asynccnt 0" ::: "memory");
# endif
#endif
    }
    __syncthreads();                        // everyone's copies for 'cur' are done

    __builtin_prefetch(At + (size_t)(lane & 15) * D_DIM + k0 + 64, 0, 1);
    v16bf a = load_frag_g(At, k0, lane);

    // hoist all LDS fragment reads so DS waits can be partial: WMMA t overlaps
    // with the ds_load_b128s of tiles t+1..7 instead of a full dscnt drain each.
    v16bf b[8];
#pragma unroll
    for (int t = 0; t < 8; ++t) b[t] = load_frag_lds(cur, tb + t, lane);
#pragma unroll
    for (int t = 0; t < 8; ++t) {
      acc[t] = __builtin_amdgcn_wmma_f32_16x16x32_bf16(false, a, false, b[t],
                                                       (short)0, acc[t], false, false);
    }
    __syncthreads();                        // done reading 'cur' before it is overwritten
  }

  // C/D layout: VGPR r, lanes 0-15 -> (M=r, N=lane); lanes 16-31 -> (M=r+8, N=lane-16)
  const int ncol  = lane & 15;
  const int rbase = (lane >> 4) * 8;
  const int nw0   = n0 + wn * 128;
#pragma unroll
  for (int t = 0; t < 8; ++t) {
#pragma unroll
    for (int r = 0; r < 8; ++r) {
      sim[(size_t)(m0 + rbase + r) * Q_SZ + nw0 + t * 16 + ncol] = acc[t][r] * INV_TEMP;
    }
  }
}

// ---------- kernel 4: per-level anchor keys ----------

__global__ void hmlc_keys(const int* __restrict__ labels, const int* __restrict__ base_loc,
                          long long* __restrict__ ka, int ncols) {
  int i = blockIdx.x * blockDim.x + threadIdx.x;
  if (i >= B_ANCH) return;
  long long base = (long long)(*base_loc) + 1;
  long long pw0 = base * base * base;
  long long pw1 = (ncols > 1) ? base * base : 0;
  long long pw2 = (ncols > 2) ? base : 0;
  ka[i] = (long long)labels[i * L_LAB + 0] * pw0 +
          (long long)labels[i * L_LAB + 1] * pw1 +
          (long long)labels[i * L_LAB + 2] * pw2;
}

// ---------- kernel 5: one block per anchor row: online masked softmax stats ----------

__global__ __launch_bounds__(256) void hmlc_row(const float* __restrict__ sim,
                                                const int* __restrict__ labels_q,
                                                const int* __restrict__ qa,
                                                const int* __restrict__ aa,
                                                const long long* __restrict__ ka,
                                                const int* __restrict__ base_loc,
                                                float* __restrict__ rowstats, int ncols) {
  const int i = blockIdx.x;
  long long base = (long long)(*base_loc) + 1;
  long long pw0 = base * base * base;
  long long pw1 = (ncols > 1) ? base * base : 0;
  long long pw2 = (ncols > 2) ? base : 0;
  const long long kai = ka[i];
  const int aai = aa[i];
  const float* row = sim + (size_t)i * Q_SZ;

  float m = -INFINITY, s = 0.0f, sp = 0.0f, c = 0.0f;
  for (int j = threadIdx.x; j < Q_SZ; j += 256) {
    if (!qa[j]) continue;
    float x = row[j];
    if (x > m) { s = s * __expf(m - x) + 1.0f; m = x; }
    else       { s += __expf(x - m); }
    if (aai) {
      long long kq = (long long)labels_q[j * L_LAB + 0] * pw0 +
                     (long long)labels_q[j * L_LAB + 1] * pw1 +
                     (long long)labels_q[j * L_LAB + 2] * pw2;
      if (kq == kai) { sp += x; c += 1.0f; }
    }
  }

  __shared__ float sm[256], ss[256], ssp[256], scnt[256];
  const int t = threadIdx.x;
  sm[t] = m; ss[t] = s; ssp[t] = sp; scnt[t] = c;
  __syncthreads();
  for (int off = 128; off > 0; off >>= 1) {
    if (t < off) {
      float m1 = sm[t], s1 = ss[t], m2 = sm[t + off], s2 = ss[t + off];
      if (s1 == 0.0f)      { sm[t] = m2; ss[t] = s2; }
      else if (s2 != 0.0f) {
        if (m2 > m1) { ss[t] = s1 * __expf(m1 - m2) + s2; sm[t] = m2; }
        else         { ss[t] = s1 + s2 * __expf(m2 - m1); }
      }
      ssp[t]  += ssp[t + off];
      scnt[t] += scnt[t + off];
    }
    __syncthreads();
  }
  if (t == 0) {
    rowstats[i * 4 + 0] = sm[0];
    rowstats[i * 4 + 1] = ss[0];
    rowstats[i * 4 + 2] = ssp[0];
    rowstats[i * 4 + 3] = scnt[0];
  }
}

// ---------- kernel 6: deactivate queue entries matched by (pre-dedup) anchors ----------

__global__ void hmlc_queue(const int* __restrict__ labels_q, const long long* __restrict__ ka,
                           const int* __restrict__ aa, int* __restrict__ qa,
                           const int* __restrict__ base_loc, int ncols) {
  int j = blockIdx.x * blockDim.x + threadIdx.x;
  if (j >= Q_SZ || !qa[j]) return;
  long long base = (long long)(*base_loc) + 1;
  long long pw0 = base * base * base;
  long long pw1 = (ncols > 1) ? base * base : 0;
  long long pw2 = (ncols > 2) ? base : 0;
  long long kq = (long long)labels_q[j * L_LAB + 0] * pw0 +
                 (long long)labels_q[j * L_LAB + 1] * pw1 +
                 (long long)labels_q[j * L_LAB + 2] * pw2;
  for (int i = 0; i < B_ANCH; ++i)
    if (aa[i] && ka[i] == kq) { qa[j] = 0; return; }
}

// ---------- kernel 7: layer loss + hmce accumulation + anchor dedup ----------

__global__ __launch_bounds__(1024) void hmlc_finalize(const float* __restrict__ rowstats,
                                                      float* __restrict__ scal,
                                                      int* __restrict__ aa,
                                                      int* __restrict__ order,
                                                      const long long* __restrict__ ka,
                                                      float coef, int do_update,
                                                      float* __restrict__ out, int is_last) {
  const int i = threadIdx.x;  // 1024 threads == B_ANCH
  float lm = rowstats[i * 4 + 0];
  float s  = rowstats[i * 4 + 1];
  float sp = rowstats[i * 4 + 2];
  float c  = rowstats[i * 4 + 3];
  float loss = 0.0f, has = 0.0f;
  if (c > 0.0f) {
    float mlpp = (sp - c * lm - c * (logf(s) + 1e-12f)) / (c + 1e-12f);
    loss = -mlpp;  // TEMP/BASE_TEMP == 1
    has = 1.0f;
  }

  __shared__ float rl[1024], rn[1024];
  rl[i] = loss; rn[i] = has;
  __syncthreads();
  for (int off = 512; off > 0; off >>= 1) {
    if (i < off) { rl[i] += rl[i + off]; rn[i] += rn[i + off]; }
    __syncthreads();
  }
  if (i == 0) {
    float layer = rl[0] / (rn[0] + 1e-12f);
    float ml  = scal[1];
    float ll  = fmaxf(ml, layer);
    float cum = scal[0] + coef * ll;
    scal[0] = cum;
    scal[1] = ll;
    if (is_last) out[0] = cum;
  }
  if (!do_update) return;

  __shared__ long long ska[1024];
  __shared__ int sord[1024], sact[1024], skept[1024];
  ska[i] = ka[i]; sord[i] = order[i]; sact[i] = aa[i];
  __syncthreads();
  int kept = 0;
  if (sact[i]) {
    long long ki = ska[i];
    int mo = -1;
    for (int j = 0; j < B_ANCH; ++j)
      if (sact[j] && ska[j] == ki && sord[j] > mo) mo = sord[j];
    kept = (sord[i] == mo);
  }
  skept[i] = kept;
  __syncthreads();
  int rank = 0;
  if (kept) {
    long long ki = ska[i];
    for (int j = 0; j < B_ANCH; ++j)
      if (skept[j] && ska[j] < ki) rank++;
  }
  aa[i] = kept;
  order[i] = kept ? rank : -1;
}

// ---------- host ----------

extern "C" void kernel_launch(void* const* d_in, const int* in_sizes, int n_in,
                              void* d_out, int out_size, void* d_ws, size_t ws_size,
                              hipStream_t stream) {
  (void)in_sizes; (void)n_in; (void)out_size; (void)ws_size;
  const float* feats   = (const float*)d_in[0];
  const int*   labels  = (const int*)d_in[1];
  const float* featsq  = (const float*)d_in[2];
  const int*   labelsq = (const int*)d_in[3];
  float* out = (float*)d_out;

  char* p = (char*)d_ws;
  float* sim = (float*)p;            p += (size_t)B_ANCH * Q_SZ * sizeof(float);
  u16* abf = (u16*)p;                p += (size_t)B_ANCH * D_DIM * sizeof(u16);
  u16* bbf = (u16*)p;                p += (size_t)Q_SZ * D_DIM * sizeof(u16);
  float* rowstats = (float*)p;       p += (size_t)B_ANCH * 4 * sizeof(float);
  long long* ka = (long long*)p;     p += (size_t)B_ANCH * sizeof(long long);
  int* aa = (int*)p;                 p += (size_t)B_ANCH * sizeof(int);
  int* order = (int*)p;              p += (size_t)B_ANCH * sizeof(int);
  int* qa = (int*)p;                 p += (size_t)Q_SZ * sizeof(int);
  int* base_loc = (int*)p;           p += sizeof(int);
  p = (char*)(((uintptr_t)p + 15) & ~(uintptr_t)15);
  float* scal = (float*)p;           // [0]=cum, [1]=max_lower

  hmlc_init<<<dim3((Q_SZ + 255) / 256), dim3(256), 0, stream>>>(qa, aa, order, base_loc, scal);

  const int tot = (B_ANCH + Q_SZ) * L_LAB;
  hmlc_base<<<dim3((tot + 255) / 256), dim3(256), 0, stream>>>(labels, labelsq, base_loc);

  {
    size_t na = (size_t)B_ANCH * D_DIM;
    size_t nb = (size_t)Q_SZ * D_DIM;
    hmlc_cvt<<<dim3((unsigned)((na / 8 + 255) / 256)), dim3(256), 0, stream>>>(feats, abf, na);
    hmlc_cvt<<<dim3((unsigned)((nb / 8 + 255) / 256)), dim3(256), 0, stream>>>(featsq, bbf, nb);
  }

  hmlc_gemm<<<dim3(Q_SZ / 512, B_ANCH / 32), dim3(256), 2 * SLAB_B, stream>>>(abf, bbf, sim);

  const float coefs[3] = {2.0f, 1.4142135623730951f, 1.2599210498948732f};
  for (int l = 1; l <= 3; ++l) {
    const int ncols = L_LAB - l;
    hmlc_keys<<<dim3((B_ANCH + 255) / 256), dim3(256), 0, stream>>>(labels, base_loc, ka, ncols);
    hmlc_row<<<dim3(B_ANCH), dim3(256), 0, stream>>>(sim, labelsq, qa, aa, ka, base_loc, rowstats, ncols);
    if (l < 3)
      hmlc_queue<<<dim3(Q_SZ / 256), dim3(256), 0, stream>>>(labelsq, ka, aa, qa, base_loc, ncols);
    hmlc_finalize<<<dim3(1), dim3(1024), 0, stream>>>(rowstats, scal, aa, order, ka,
                                                      coefs[l - 1], (l < 3) ? 1 : 0,
                                                      out, (l == 3) ? 1 : 0);
  }
}
